// Decoder_82403242541520
// MI455X (gfx1250) — matrix operand order repaired and checked by reading, not compiled, not run on hardware
//
#include <hip/hip_runtime.h>
#include <hip/hip_bf16.h>
#include <stdint.h>

// Problem dims (fixed by the reference)
#define B_ 64
#define T_ 256
#define S_ 256
#define E_ 256
#define H_ 1024

typedef __attribute__((ext_vector_type(16))) __bf16 v16bf;
typedef __attribute__((ext_vector_type(8)))  float  v8f;

union FragAB { unsigned int u[8]; v16bf v; };
union FragC  { float f[8]; v8f v; };

__device__ __forceinline__ unsigned short f2bf(float x) {
  unsigned int u = __float_as_uint(x);
  unsigned int r = (u + 0x7FFFu + ((u >> 16) & 1u)) >> 16;  // round-to-nearest-even
  return (unsigned short)r;
}
__device__ __forceinline__ float bf2f(unsigned short u) {
  return __uint_as_float((unsigned int)u << 16);
}

// ---------------------------------------------------------------------------
// f32 -> bf16 bulk conversion
// ---------------------------------------------------------------------------
__global__ __launch_bounds__(256) void cvt_f32_bf16(const float* __restrict__ src,
                                                    unsigned short* __restrict__ dst,
                                                    int n) {
  int i = blockIdx.x * 256 + threadIdx.x;
  if (i < n) dst[i] = f2bf(src[i]);
}

// ---------------------------------------------------------------------------
// Fragment loaders (per CDNA5 WMMA VGPR layouts, 16-bit operands)
// ---------------------------------------------------------------------------
__device__ __forceinline__ void load_fragA(FragAB& a, const unsigned short* aRow,
                                           int k0, int half) {
#pragma unroll
  for (int v = 0; v < 8; ++v) {
    int kk = (v < 4) ? (v * 2 + half * 8) : (16 + (v - 4) * 2 + half * 8);
    a.u[v] = *(const unsigned int*)(const void*)(aRow + k0 + kk);
  }
}
__device__ __forceinline__ void load_fragB(FragAB& b, const unsigned short* bRow,
                                           int k0, int half) {
#pragma unroll
  for (int v = 0; v < 8; ++v) {
    int kk = (v < 4) ? (v * 2 + half * 8) : (16 + (v - 4) * 2 + half * 8);
    b.u[v] = *(const unsigned int*)(const void*)(bRow + k0 + kk);
  }
}

// ---------------------------------------------------------------------------
// WMMA GEMM (latency flavor): C[M,N] (+)= A[M,K]*Bt[N,K]^T (+bias).
// 1 wave = one 16x16 tile; block = 8 waves (2M x 4N) = 32x64 tile.
// Grid = (N/64, M/32). Used for the small-M per-step GEMMs (max parallelism).
// ---------------------------------------------------------------------------
__global__ __launch_bounds__(256) void wmma_gemm(
    const unsigned short* __restrict__ A, int lda,
    const unsigned short* __restrict__ Bt, int ldb,
    float* __restrict__ C, int ldc,
    const float* __restrict__ bias, int accumulate, int K)
{
  const int lane = threadIdx.x & 31;
  const int wave = threadIdx.x >> 5;
  const int half = lane >> 4;
  const int l16  = lane & 15;
  const int row0 = blockIdx.y * 32 + (wave >> 2) * 16;
  const int col0 = blockIdx.x * 64 + (wave & 3) * 16;

  const unsigned short* aRow = A  + (size_t)(row0 + l16) * lda;
  const unsigned short* bRow = Bt + (size_t)(col0 + l16) * ldb;

  v8f acc = {};
  FragC cfr;
  if (accumulate) {
#pragma unroll
    for (int v = 0; v < 8; ++v)
      cfr.f[v] = C[(size_t)(row0 + v + half * 8) * ldc + col0 + l16];
    acc = cfr.v;
  }

  for (int k0 = 0; k0 < K; k0 += 32) {
    FragAB a, b;
    load_fragA(a, aRow, k0, half);
    load_fragB(b, bRow, k0, half);
    acc = __builtin_amdgcn_wmma_f32_16x16x32_bf16(
        false, a.v, false, b.v, (short)0, acc, false, false);
  }

  float bval = bias ? bias[col0 + l16] : 0.0f;
  cfr.v = acc;
#pragma unroll
  for (int v = 0; v < 8; ++v)
    C[(size_t)(row0 + v + half * 8) * ldc + col0 + l16] = cfr.f[v] + bval;
}

// ---------------------------------------------------------------------------
// WMMA GEMM (throughput flavor, 2x2 register-blocked): each wave computes a
// 32x32 tile (4 WMMAs per 32-K step, 2 b128 loads per WMMA). Block = 8 waves
// (2M x 4N) = 64x128 tile. Grid = (N/128, M/64).
// Optional bf16 output (Cbf != null) for L2-resident activation tensors.
// ---------------------------------------------------------------------------
__global__ __launch_bounds__(256) void wmma_gemm_rb(
    const unsigned short* __restrict__ A, int lda,
    const unsigned short* __restrict__ Bt, int ldb,
    float* __restrict__ C, int ldc,
    unsigned short* __restrict__ Cbf,
    const float* __restrict__ bias, int accumulate, int K)
{
  const int lane = threadIdx.x & 31;
  const int wave = threadIdx.x >> 5;
  const int half = lane >> 4;
  const int l16  = lane & 15;
  const int row0 = blockIdx.y * 64 + (wave >> 2) * 32;
  const int col0 = blockIdx.x * 128 + (wave & 3) * 32;

  const unsigned short* aRow0 = A  + (size_t)(row0 + l16) * lda;
  const unsigned short* aRow1 = aRow0 + (size_t)16 * lda;
  const unsigned short* bRow0 = Bt + (size_t)(col0 + l16) * ldb;
  const unsigned short* bRow1 = bRow0 + (size_t)16 * ldb;

  v8f acc00 = {}, acc01 = {}, acc10 = {}, acc11 = {};
  FragC cfr;
  if (accumulate) {
#pragma unroll
    for (int v = 0; v < 8; ++v) cfr.f[v] = C[(size_t)(row0 +      v + half*8) * ldc + col0      + l16];
    acc00 = cfr.v;
#pragma unroll
    for (int v = 0; v < 8; ++v) cfr.f[v] = C[(size_t)(row0 +      v + half*8) * ldc + col0 + 16 + l16];
    acc01 = cfr.v;
#pragma unroll
    for (int v = 0; v < 8; ++v) cfr.f[v] = C[(size_t)(row0 + 16 + v + half*8) * ldc + col0      + l16];
    acc10 = cfr.v;
#pragma unroll
    for (int v = 0; v < 8; ++v) cfr.f[v] = C[(size_t)(row0 + 16 + v + half*8) * ldc + col0 + 16 + l16];
    acc11 = cfr.v;
  }

  for (int k0 = 0; k0 < K; k0 += 32) {
    FragAB a0, a1, b0, b1;
    load_fragA(a0, aRow0, k0, half);
    load_fragA(a1, aRow1, k0, half);
    load_fragB(b0, bRow0, k0, half);
    load_fragB(b1, bRow1, k0, half);
    acc00 = __builtin_amdgcn_wmma_f32_16x16x32_bf16(false, a0.v, false, b0.v, (short)0, acc00, false, false);
    acc01 = __builtin_amdgcn_wmma_f32_16x16x32_bf16(false, a0.v, false, b1.v, (short)0, acc01, false, false);
    acc10 = __builtin_amdgcn_wmma_f32_16x16x32_bf16(false, a1.v, false, b0.v, (short)0, acc10, false, false);
    acc11 = __builtin_amdgcn_wmma_f32_16x16x32_bf16(false, a1.v, false, b1.v, (short)0, acc11, false, false);
  }

  float bv0 = bias ? bias[col0 + l16] : 0.0f;
  float bv1 = bias ? bias[col0 + 16 + l16] : 0.0f;

  if (Cbf) {  // bf16 output path (no accumulate/bias users need this)
    cfr.v = acc00;
#pragma unroll
    for (int v = 0; v < 8; ++v) Cbf[(size_t)(row0 +      v + half*8) * ldc + col0      + l16] = f2bf(cfr.f[v] + bv0);
    cfr.v = acc01;
#pragma unroll
    for (int v = 0; v < 8; ++v) Cbf[(size_t)(row0 +      v + half*8) * ldc + col0 + 16 + l16] = f2bf(cfr.f[v] + bv1);
    cfr.v = acc10;
#pragma unroll
    for (int v = 0; v < 8; ++v) Cbf[(size_t)(row0 + 16 + v + half*8) * ldc + col0      + l16] = f2bf(cfr.f[v] + bv0);
    cfr.v = acc11;
#pragma unroll
    for (int v = 0; v < 8; ++v) Cbf[(size_t)(row0 + 16 + v + half*8) * ldc + col0 + 16 + l16] = f2bf(cfr.f[v] + bv1);
  } else {
    cfr.v = acc00;
#pragma unroll
    for (int v = 0; v < 8; ++v) C[(size_t)(row0 +      v + half*8) * ldc + col0      + l16] = cfr.f[v] + bv0;
    cfr.v = acc01;
#pragma unroll
    for (int v = 0; v < 8; ++v) C[(size_t)(row0 +      v + half*8) * ldc + col0 + 16 + l16] = cfr.f[v] + bv1;
    cfr.v = acc10;
#pragma unroll
    for (int v = 0; v < 8; ++v) C[(size_t)(row0 + 16 + v + half*8) * ldc + col0      + l16] = cfr.f[v] + bv0;
    cfr.v = acc11;
#pragma unroll
    for (int v = 0; v < 8; ++v) C[(size_t)(row0 + 16 + v + half*8) * ldc + col0 + 16 + l16] = cfr.f[v] + bv1;
  }
}

// ---------------------------------------------------------------------------
// Bridge: h0 = tanh(enc_h_final @ W_bridge^T + b_bridge) -> bf16 (feeds GEMM)
//         c0 = tanh(enc_c_final @ W_bridge^T + b_bridge) -> f32 state
// ---------------------------------------------------------------------------
__global__ __launch_bounds__(256) void bridge_kernel(
    const float* __restrict__ ench, const float* __restrict__ encc,
    const float* __restrict__ Wb, const float* __restrict__ bb,
    unsigned short* __restrict__ h0_bf, float* __restrict__ c_state)
{
  int idx = blockIdx.x * 256 + threadIdx.x;   // [0, 2*B*H)
  int which = idx / (B_ * H_);
  int rem = idx % (B_ * H_);
  int b = rem / H_, n = rem % H_;
  const float* src = (which == 0) ? (ench + (size_t)b * 2 * H_)
                                  : (encc + (size_t)b * 2 * H_);
  const float* w = Wb + (size_t)n * 2 * H_;
  float s = bb[n];
  for (int k = 0; k < 2 * H_; ++k) s += src[k] * w[k];
  float t = tanhf(s);
  if (which == 0) h0_bf[rem] = f2bf(t);
  else            c_state[rem] = t;
}

// ---------------------------------------------------------------------------
// LSTM pointwise: gates[B,4H] (i|f|g|o) -> update c, produce h.
// ---------------------------------------------------------------------------
__global__ __launch_bounds__(256) void lstm_pointwise(
    const float* __restrict__ gates,
    float* __restrict__ c_state,
    float* __restrict__ hseq_out,
    unsigned short* __restrict__ hseq_bf,
    float* __restrict__ hfinal,
    int t)
{
  int idx = blockIdx.x * 256 + threadIdx.x;   // B*H
  int b = idx / H_, n = idx % H_;
  const float* g = gates + (size_t)b * 4 * H_;
  float gi = g[n], gf = g[H_ + n], gg = g[2 * H_ + n], go = g[3 * H_ + n];
  float si = 1.f / (1.f + __expf(-gi));
  float sf = 1.f / (1.f + __expf(-gf));
  float so = 1.f / (1.f + __expf(-go));
  float c = sf * c_state[idx] + si * tanhf(gg);
  c_state[idx] = c;
  float h = so * tanhf(c);
  size_t row = (size_t)b * T_ + t;
  hseq_out[row * H_ + n] = h;
  hseq_bf[row * H_ + n] = f2bf(h);
  if (hfinal) hfinal[idx] = h;
}

// ---------------------------------------------------------------------------
// energy[b,s] = v . tanh(q[b] + proj_key_bf[b,s]), masked.  One block per
// (b,s); proj_key is bf16 so the 256 per-step sweeps stay L2-resident.
// ---------------------------------------------------------------------------
__global__ __launch_bounds__(256) void attn_energy(
    const float* __restrict__ q,
    const unsigned short* __restrict__ proj_key_bf,
    const float* __restrict__ v_energy,
    const unsigned char* __restrict__ mask,
    float* __restrict__ energy)
{
  __shared__ float red[256];
  int bs = blockIdx.x;
  int b = bs / S_;
  const float* qb = q + (size_t)b * H_;
  const unsigned short* pk = proj_key_bf + (size_t)bs * H_;
  float p = 0.f;
  for (int h = threadIdx.x; h < H_; h += 256)
    p += tanhf(qb[h] + bf2f(pk[h])) * v_energy[h];
  red[threadIdx.x] = p;
  __syncthreads();
  for (int st = 128; st > 0; st >>= 1) {
    if (threadIdx.x < st) red[threadIdx.x] += red[threadIdx.x + st];
    __syncthreads();
  }
  if (threadIdx.x == 0)
    energy[bs] = mask[bs] ? red[0] : -1e9f;
}

// ---------------------------------------------------------------------------
// softmax over S per batch + ctx = probs @ encoder_hidden (bf16, L2-resident)
// -> ctx_bf[b*T+t].  One block per b; S_ == blockDim.
// ---------------------------------------------------------------------------
__global__ __launch_bounds__(256) void softmax_ctx(
    const float* __restrict__ energy,
    const unsigned short* __restrict__ enc_bf,   // bf16 [B*S, 2H]
    unsigned short* __restrict__ ctx_bf,         // [B*T, 2H]
    int t)
{
  __shared__ float prob[S_];
  __shared__ float red[S_];
  int b = blockIdx.x;
  int s = threadIdx.x;
  float e = energy[b * S_ + s];
  red[s] = e; __syncthreads();
  for (int st = 128; st > 0; st >>= 1) {
    if (s < st) red[s] = fmaxf(red[s], red[s + st]);
    __syncthreads();
  }
  float m = red[0]; __syncthreads();
  float ex = __expf(e - m);
  red[s] = ex; __syncthreads();
  for (int st = 128; st > 0; st >>= 1) {
    if (s < st) red[s] += red[s + st];
    __syncthreads();
  }
  prob[s] = ex / red[0];
  __syncthreads();
  const unsigned short* encb = enc_bf + (size_t)b * S_ * 2 * H_;
  size_t crow = ((size_t)b * T_ + t) * 2 * H_;
  for (int k = threadIdx.x; k < 2 * H_; k += 256) {
    float acc = 0.f;
    for (int ss = 0; ss < S_; ++ss)
      acc += prob[ss] * bf2f(encb[(size_t)ss * 2 * H_ + k]);
    ctx_bf[crow + k] = f2bf(acc);
  }
}

// ---------------------------------------------------------------------------
extern "C" void kernel_launch(void* const* d_in, const int* in_sizes, int n_in,
                              void* d_out, int out_size, void* d_ws, size_t ws_size,
                              hipStream_t stream)
{
  (void)in_sizes; (void)n_in; (void)out_size; (void)ws_size;

  const float* trg_embed        = (const float*)d_in[0];
  const float* enc_hidden       = (const float*)d_in[1];
  const float* ench_final       = (const float*)d_in[2];
  const float* encc_final       = (const float*)d_in[3];
  const unsigned char* src_mask = (const unsigned char*)d_in[4];
  /* d_in[5] trg_mask unused by the reference math */
  const float* W_ih    = (const float*)d_in[6];
  const float* W_hh    = (const float*)d_in[7];
  const float* b_lstm  = (const float*)d_in[8];
  const float* W_bridge= (const float*)d_in[9];
  const float* b_bridge= (const float*)d_in[10];
  const float* W_key   = (const float*)d_in[11];
  const float* W_query = (const float*)d_in[12];
  const float* v_energy= (const float*)d_in[13];
  const float* W_pre   = (const float*)d_in[14];

  char* ws = (char*)d_ws;
  size_t off = 0;
  auto alloc = [&](size_t bytes) -> void* {
    void* p = (void*)(ws + off);
    off += (bytes + 255) & ~(size_t)255;
    return p;
  };

  const int KP = E_ + 3 * H_;  // 3328, pre-projection K
  unsigned short* tr_bf   = (unsigned short*)alloc((size_t)B_ * T_ * E_ * 2);
  unsigned short* enc_bf  = (unsigned short*)alloc((size_t)B_ * S_ * 2 * H_ * 2);
  unsigned short* Wih_bf  = (unsigned short*)alloc((size_t)4 * H_ * E_ * 2);
  unsigned short* Whh_bf  = (unsigned short*)alloc((size_t)4 * H_ * H_ * 2);
  unsigned short* Wkey_bf = (unsigned short*)alloc((size_t)H_ * 2 * H_ * 2);
  unsigned short* Wq_bf   = (unsigned short*)alloc((size_t)H_ * H_ * 2);
  unsigned short* Wpre_bf = (unsigned short*)alloc((size_t)H_ * KP * 2);
  unsigned short* pk_bf   = (unsigned short*)alloc((size_t)B_ * S_ * H_ * 2);
  unsigned short* hseq_bf = (unsigned short*)alloc((size_t)B_ * T_ * H_ * 2);
  unsigned short* ctx_bf  = (unsigned short*)alloc((size_t)B_ * T_ * 2 * H_ * 2);
  unsigned short* h0_bf   = (unsigned short*)alloc((size_t)B_ * H_ * 2);
  float* c_state          = (float*)alloc((size_t)B_ * H_ * 4);
  float* q_buf            = (float*)alloc((size_t)B_ * H_ * 4);
  float* gates            = (float*)alloc((size_t)B_ * 4 * H_ * 4);
  float* energy           = (float*)alloc((size_t)B_ * S_ * 4);

  float* out_hseq   = (float*)d_out;                       // (B,T,H)
  float* out_hfinal = out_hseq + (size_t)B_ * T_ * H_;     // (1,B,H)
  float* out_pre    = out_hfinal + (size_t)B_ * H_;        // (B,T,H)

  auto cvt = [&](const float* s, unsigned short* d, size_t n) {
    cvt_f32_bf16<<<dim3((unsigned)((n + 255) / 256)), dim3(256), 0, stream>>>(s, d, (int)n);
  };
  cvt(trg_embed,  tr_bf,   (size_t)B_ * T_ * E_);
  cvt(enc_hidden, enc_bf,  (size_t)B_ * S_ * 2 * H_);
  cvt(W_ih,       Wih_bf,  (size_t)4 * H_ * E_);
  cvt(W_hh,       Whh_bf,  (size_t)4 * H_ * H_);
  cvt(W_key,      Wkey_bf, (size_t)H_ * 2 * H_);
  cvt(W_query,    Wq_bf,   (size_t)H_ * H_);
  cvt(W_pre,      Wpre_bf, (size_t)H_ * KP);

  // h0 (bf16) and c0 (f32 state)
  bridge_kernel<<<dim3(2 * B_ * H_ / 256), 256, 0, stream>>>(
      ench_final, encc_final, W_bridge, b_bridge, h0_bf, c_state);

  // proj_key(bf16) = encoder_hidden @ W_key^T  (M=B*S, N=H, K=2H) — throughput GEMM
  wmma_gemm_rb<<<dim3(H_ / 128, (B_ * S_) / 64), 256, 0, stream>>>(
      enc_bf, 2 * H_, Wkey_bf, 2 * H_, nullptr, H_, pk_bf, nullptr, 0, 2 * H_);

  // ---- sequential decoder scan (latency GEMM flavor for M=64) ----
  for (int t = 0; t < T_; ++t) {
    // gates = x_t @ W_ih^T + b_lstm
    wmma_gemm<<<dim3(4 * H_ / 64, B_ / 32), 256, 0, stream>>>(
        tr_bf + (size_t)t * E_, T_ * E_, Wih_bf, E_, gates, 4 * H_,
        b_lstm, /*acc=*/0, E_);
    // gates += h_prev @ W_hh^T
    const unsigned short* hprev = (t == 0) ? h0_bf : (hseq_bf + (size_t)(t - 1) * H_);
    int ldh = (t == 0) ? H_ : T_ * H_;
    wmma_gemm<<<dim3(4 * H_ / 64, B_ / 32), 256, 0, stream>>>(
        hprev, ldh, Whh_bf, H_, gates, 4 * H_, nullptr, /*acc=*/1, H_);
    // c,h update (writes h into d_out h_seq + bf16 copy)
    lstm_pointwise<<<dim3(B_ * H_ / 256), 256, 0, stream>>>(
        gates, c_state, out_hseq, hseq_bf,
        (t == T_ - 1) ? out_hfinal : nullptr, t);
    // q = h @ W_query^T
    wmma_gemm<<<dim3(H_ / 64, B_ / 32), 256, 0, stream>>>(
        hseq_bf + (size_t)t * H_, T_ * H_, Wq_bf, H_, q_buf, H_,
        nullptr, /*acc=*/0, H_);
    // attention energy + softmax + context (bf16 L2-resident streams)
    attn_energy<<<dim3(B_ * S_), 256, 0, stream>>>(
        q_buf, pk_bf, v_energy, src_mask, energy);
    softmax_ctx<<<dim3(B_), 256, 0, stream>>>(energy, enc_bf, ctx_bf, t);
  }

  // pre = [trg_embed | h_seq | ctx_seq] @ W_pre^T, 3 K-segment passes (rb GEMM)
  wmma_gemm_rb<<<dim3(H_ / 128, (B_ * T_) / 64), 256, 0, stream>>>(
      tr_bf, E_, Wpre_bf, KP, out_pre, H_, nullptr, nullptr, /*acc=*/0, E_);
  wmma_gemm_rb<<<dim3(H_ / 128, (B_ * T_) / 64), 256, 0, stream>>>(
      hseq_bf, H_, Wpre_bf + E_, KP, out_pre, H_, nullptr, nullptr, /*acc=*/1, H_);
  wmma_gemm_rb<<<dim3(H_ / 128, (B_ * T_) / 64), 256, 0, stream>>>(
      ctx_bf, 2 * H_, Wpre_bf + (E_ + H_), KP, out_pre, H_, nullptr, nullptr, /*acc=*/1, 2 * H_);
}
